// KmeansModel_33191507264089
// MI455X (gfx1250) — matrix operand order repaired and checked by reading, not compiled, not run on hardware
//
#include <hip/hip_runtime.h>
#include <hip/hip_bf16.h>

// ---------------------------------------------------------------------------
// K-means assignment on MI455X (gfx1250, wave32, WMMA):
//   assigned[n] = argmin_k |x_n - C_k|^2 = argmax_k (x_n . C_k - 0.5*|C_k|^2)
// GEMM 32768x1024 @ 1024x4096 fused with row-argmax.
// fp32 emulated via bf16 hi/lo split: x.C ~= hi.Hi + hi.Lo + lo.Hi
// (3x v_wmma_f32_16x16x32_bf16 per K-chunk of 32, fp32 accumulate).
// Each wave computes a 16x32 output slab with TWO independent accumulators
// so the XDL pipe always has independent WMMA chains in flight.
// ---------------------------------------------------------------------------

typedef __bf16 bf16_t;
typedef __attribute__((ext_vector_type(16))) __bf16 v16bf;
typedef __attribute__((ext_vector_type(8)))  float  v8f;

#define D_DIM 1024   // reduction dim
#define K_DIM 4096   // centroids
#define N_ROWS 32768 // B*T
#define MBLK 32      // rows per workgroup
#define KTILES (K_DIM / 16)        // 256 column tiles of 16
#define DCHUNKS (D_DIM / 32)       // 32 reduction chunks of 32
// B-fragment tile in workspace: 32(d) x 16(k) bf16, lane-major, 1024 bytes.
#define TILE_ELEMS 512
#define TILE_STRIDE ((size_t)DCHUNKS * TILE_ELEMS)   // elems per k-tile column

// ---------------------------------------------------------------------------
// Pre-pass: split fp32 C[d,k] into bf16 hi/lo, pre-swizzled into WMMA
// B-fragment layout so the GEMM wave loads 32 contiguous bytes per lane.
// B 32x16 bf16, contiguous k order used here (same as A): lane = n + 16*(kk/16), element e = kk%16.
// ---------------------------------------------------------------------------
__global__ __launch_bounds__(256) void kmeans_convertC_kernel(
    const float* __restrict__ C, bf16_t* __restrict__ Bhi,
    bf16_t* __restrict__ Blo) {
  int idx = blockIdx.x * 256 + threadIdx.x;     // coalesced read along k
  int d = idx >> 12;                            // / 4096
  int k = idx & (K_DIM - 1);
  float v = C[idx];
  __bf16 h = (__bf16)v;
  __bf16 l = (__bf16)(v - (float)h);
  int dt = d >> 5, kk = d & 31;                 // d-chunk, offset in chunk
  int kt = k >> 4, n  = k & 15;                 // k-tile, column in tile
  int lane = n + ((kk >> 4) << 4);              // n + 16*(kk/16)
  int e    = kk & 15;
  size_t dst = (size_t)(kt * DCHUNKS + dt) * TILE_ELEMS + lane * 16 + e;
  Bhi[dst] = h;
  Blo[dst] = l;
}

// ---------------------------------------------------------------------------
// Fused GEMM + argmax.  256 threads = 8 waves, arranged 2 (M) x 4 (N);
// each wave owns two adjacent 16-col tiles (16x32 output, 2 accumulators).
// LDS: A hi/lo fragments (128 KB) + reduction scratch (16 KB) = 144 KB
// (dynamic; <=160 KB so two workgroups can co-reside per 320 KB WGP).
// ---------------------------------------------------------------------------
__global__ __launch_bounds__(256) void kmeans_assign_kernel(
    const float* __restrict__ x, const float* __restrict__ Cnorm,
    const bf16_t* __restrict__ Bhi, const bf16_t* __restrict__ Blo,
    int* __restrict__ out) {
  extern __shared__ char smem[];
  bf16_t* aHi = (bf16_t*)smem;                       // [2][32][32][16] bf16
  bf16_t* aLo = aHi + 2 * DCHUNKS * 32 * 16;         // 64 KB each
  float*  redV = (float*)(aLo + 2 * DCHUNKS * 32 * 16);  // [32][64]
  int*    redI = (int*)(redV + MBLK * 64);               // [32][64]

  const int tid = threadIdx.x;
  const int blk = blockIdx.x;

  // ---- Stage x block (32 rows x 1024 d, fp32) into LDS as hi/lo bf16 in
  // A-fragment order.  A 16x32 bf16, contiguous k order used here (same as B): lane m + 16*g with
  // g = kk>>4, element e = kk&15.
  const float* xblk = x + (size_t)blk * MBLK * D_DIM;
  for (int eidx = tid; eidx < MBLK * D_DIM; eidx += 256) {
    int m = eidx >> 10;
    int d = eidx & (D_DIM - 1);
    float v = xblk[eidx];
    __bf16 h = (__bf16)v;
    __bf16 l = (__bf16)(v - (float)h);
    int wm = m >> 4, lm = m & 15;
    int dc = d >> 5, kk = d & 31;
    int lane = lm + ((kk >> 4) << 4);
    int e = kk & 15;
    int addr = (((wm * DCHUNKS + dc) << 5) + lane) * 16 + e;
    aHi[addr] = h;
    aLo[addr] = l;
  }
  __syncthreads();

  const int wave = tid >> 5;
  const int lane = tid & 31;
  const int wm = wave >> 2;      // 0..1 : 16-row band
  const int wn = wave & 3;       // 0..3 : column-tile-pair phase
  const int nloc = lane & 15;    // column within 16-wide tile

  float bestV[8];
  int   bestI[8];
#pragma unroll
  for (int r = 0; r < 8; ++r) { bestV[r] = -3.402823466e38f; bestI[r] = 0; }

  const bf16_t* aHiW = aHi + wm * DCHUNKS * 32 * 16;
  const bf16_t* aLoW = aLo + wm * DCHUNKS * 32 * 16;

  for (int t = 0; t < KTILES / 8; ++t) {           // 32 iterations
    const int ktBase = (t * 4 + wn) * 2;           // this wave's tile pair
    const int k0 = ktBase << 4;
    const bf16_t* bHiT = Bhi + (size_t)ktBase * TILE_STRIDE;
    const bf16_t* bLoT = Blo + (size_t)ktBase * TILE_STRIDE;

    // Prefetch next iteration's tile pair (global_prefetch_b8 path).
    if (t + 1 < KTILES / 8) {
      __builtin_prefetch(bHiT + 8 * TILE_STRIDE, 0, 1);
      __builtin_prefetch(bLoT + 8 * TILE_STRIDE, 0, 1);
    }

    v8f acc0 = {};
    v8f acc1 = {};
#pragma unroll 2
    for (int dc = 0; dc < DCHUNKS; ++dc) {
      v16bf ah = *(const v16bf*)(aHiW + ((dc << 5) + lane) * 16);
      v16bf al = *(const v16bf*)(aLoW + ((dc << 5) + lane) * 16);
      v16bf bh0 = *(const v16bf*)(bHiT + dc * TILE_ELEMS + lane * 16);
      v16bf bl0 = *(const v16bf*)(bLoT + dc * TILE_ELEMS + lane * 16);
      v16bf bh1 = *(const v16bf*)(bHiT + TILE_STRIDE + dc * TILE_ELEMS + lane * 16);
      v16bf bl1 = *(const v16bf*)(bLoT + TILE_STRIDE + dc * TILE_ELEMS + lane * 16);
      // (neg_a, A, neg_b, B, c_mod, C, reuse_a, reuse_b)
      // Interleave the two independent accumulator chains.
      acc0 = __builtin_amdgcn_wmma_f32_16x16x32_bf16(false, ah, false, bl0,
                                                     (short)0, acc0, false, false);
      acc1 = __builtin_amdgcn_wmma_f32_16x16x32_bf16(false, ah, false, bl1,
                                                     (short)0, acc1, false, false);
      acc0 = __builtin_amdgcn_wmma_f32_16x16x32_bf16(false, al, false, bh0,
                                                     (short)0, acc0, false, false);
      acc1 = __builtin_amdgcn_wmma_f32_16x16x32_bf16(false, al, false, bh1,
                                                     (short)0, acc1, false, false);
      acc0 = __builtin_amdgcn_wmma_f32_16x16x32_bf16(false, ah, false, bh0,
                                                     (short)0, acc0, false, false);
      acc1 = __builtin_amdgcn_wmma_f32_16x16x32_bf16(false, ah, false, bh1,
                                                     (short)0, acc1, false, false);
    }

    const int kabs0 = k0 + nloc;
    const int kabs1 = kabs0 + 16;
    const float cn0 = Cnorm[kabs0];
    const float cn1 = Cnorm[kabs1];
#pragma unroll
    for (int r = 0; r < 8; ++r) {
      float s0 = acc0[r] - 0.5f * cn0;  // argmax(s) == argmin(distance)
      float s1 = acc1[r] - 0.5f * cn1;
      if (s0 > bestV[r] || (s0 == bestV[r] && kabs0 < bestI[r])) {
        bestV[r] = s0;
        bestI[r] = kabs0;
      }
      if (s1 > bestV[r] || (s1 == bestV[r] && kabs1 < bestI[r])) {
        bestV[r] = s1;
        bestI[r] = kabs1;
      }
    }
  }

  // ---- Cross-lane / cross-wave argmax reduction.
  // C/D layout: VGPR r holds row (r) for lanes 0-15, row (r+8) for 16-31.
#pragma unroll
  for (int r = 0; r < 8; ++r) {
    int m = (wm << 4) + r + ((lane >> 4) << 3);
    int slot = (wn << 4) + nloc;
    redV[m * 64 + slot] = bestV[r];
    redI[m * 64 + slot] = bestI[r];
  }
  __syncthreads();

  if (tid < MBLK) {
    float bv = -3.402823466e38f;
    int bi = 0;
    for (int s = 0; s < 64; ++s) {
      float v = redV[tid * 64 + s];
      int   i = redI[tid * 64 + s];
      if (v > bv || (v == bv && i < bi)) { bv = v; bi = i; }
    }
    out[(size_t)blk * MBLK + tid] = bi;   // int32 argmin index
  }
}

// ---------------------------------------------------------------------------
extern "C" void kernel_launch(void* const* d_in, const int* in_sizes, int n_in,
                              void* d_out, int out_size, void* d_ws,
                              size_t ws_size, hipStream_t stream) {
  const float* x     = (const float*)d_in[0];  // [8,4096,1024]
  const float* C     = (const float*)d_in[1];  // [1024,4096]
  const float* Cnorm = (const float*)d_in[2];  // [1,4096]
  int* out = (int*)d_out;                      // [8,4096] int32 indices

  // Workspace: C_hi / C_lo bf16, 8 MB each (ws must be >= 16 MB).
  bf16_t* Bhi = (bf16_t*)d_ws;
  bf16_t* Blo = Bhi + (size_t)D_DIM * K_DIM;

  kmeans_convertC_kernel<<<(D_DIM * K_DIM) / 256, 256, 0, stream>>>(C, Bhi,
                                                                    Blo);

  size_t lds_bytes = (size_t)(2 * 2 * DCHUNKS * 32 * 16) * sizeof(bf16_t) +
                     (size_t)MBLK * 64 * (sizeof(float) + sizeof(int));
  kmeans_assign_kernel<<<N_ROWS / MBLK, 256, lds_bytes, stream>>>(
      x, Cnorm, Bhi, Blo, out);
}
